// DropoutUncertaintyLSTMCell_12025908428918
// MI455X (gfx1250) — matrix operand order repaired and checked by reading, not compiled, not run on hardware
//
#include <hip/hip_runtime.h>
#include <hip/hip_bf16.h>
#include <math.h>

// CDNA5 / gfx1250, wave32. Matrix core path: V_WMMA_F32_16X16X32_BF16.
typedef __attribute__((ext_vector_type(16))) __bf16 v16bf;
typedef __attribute__((ext_vector_type(8)))  float  v8f;

constexpr int T_ = 256, B_ = 64, I_ = 512, H_ = 1024, G_ = 4;

// workspace layout (bytes); total ~269 MB
constexpr size_t WXB_OFF = 0;                                    // bf16 [G][H][I]
constexpr size_t UHB_OFF = WXB_OFF + (size_t)G_ * H_ * I_ * 2;   // bf16 [G][H][H]
constexpr size_t HMB_OFF = UHB_OFF + (size_t)G_ * H_ * H_ * 2;   // bf16 [2][G][B][H] (double buffer)
constexpr size_t C_OFF   = HMB_OFF + (size_t)2 * G_ * B_ * H_ * 2; // f32 [B][H]
constexpr size_t XP_OFF  = C_OFF + (size_t)B_ * H_ * 4;          // f32 [T][G][B][H]

__global__ void cvt_bf16(const float* __restrict__ src, __bf16* __restrict__ dst, int n) {
  int i = blockIdx.x * blockDim.x + threadIdx.x;
  if (i < n) dst[i] = (__bf16)src[i];
}

// hmb0[g][b][h] = bf16(h0[b][h] * zh[g][b][h]); c = c0
__global__ void init_state(const float* __restrict__ h0, const float* __restrict__ c0,
                           const float* __restrict__ zh,
                           __bf16* __restrict__ hmb0, float* __restrict__ c) {
  int i = blockIdx.x * blockDim.x + threadIdx.x;
  if (i < G_ * B_ * H_) {
    int bh = i % (B_ * H_);
    hmb0[i] = (__bf16)(h0[bh] * zh[i]);
    if (i < B_ * H_) c[i] = c0[i];
  }
}

// xp[t][g][b][n] = sum_k (x[t][b][k]*zx[g][b][k]) * Wx[g][n][k] + bx[g][n]
// One workgroup: one gate, 64(M=T*B rows) x 64(N) tile; 8 waves, each 16x32.
__global__ void xp_gemm(const float* __restrict__ x, const float* __restrict__ zx,
                        const __bf16* __restrict__ Wxb, const float* __restrict__ bx,
                        float* __restrict__ xp) {
  constexpr int NB = H_ / 64;        // 16
  constexpr int MB = (T_ * B_) / 64; // 256
  int bid  = blockIdx.x;
  int g    = bid / (MB * NB);
  int rem  = bid % (MB * NB);
  int mblk = rem / NB;
  int nblk = rem % NB;
  int wave = threadIdx.x >> 5;
  int lane = threadIdx.x & 31;
  int hi   = lane >> 4;              // half-wave select (ISA fragment layout)
  int lr   = lane & 15;
  int m0 = mblk * 64 + (wave >> 1) * 16;
  int n0 = nblk * 64 + (wave & 1) * 32;
  int row = m0 + lr;                 // index into flattened (t,b)
  int b   = row % B_;
  const float* xr  = x  + (size_t)row * I_;
  const float* zxr = zx + ((size_t)g * B_ + b) * I_;
  v8f acc0 = {}, acc1 = {};
  for (int kb = 0; kb < I_; kb += 32) {
    // A fragment (16x32 bf16): K = kb + 16*hi + e; mask fused on the fly
    v16bf a;
    int k0 = kb + 16 * hi;
#pragma unroll
    for (int e = 0; e < 8; ++e) {
      a[e]     = (__bf16)(xr[k0 + e]      * zxr[k0 + e]);
      a[e + 8] = (__bf16)(xr[k0 + 8 + e] * zxr[k0 + 8 + e]);
    }
    // B fragments (32x16 bf16): lane covers col n0+lr, 16 contiguous k -> one 32B load
    const v16bf b0 = *(const v16bf*)(Wxb + ((size_t)g * H_ + (n0 + lr))      * I_ + kb + 16 * hi);
    const v16bf b1 = *(const v16bf*)(Wxb + ((size_t)g * H_ + (n0 + 16 + lr)) * I_ + kb + 16 * hi);
    acc0 = __builtin_amdgcn_wmma_f32_16x16x32_bf16(false, a, false, b0, (short)0, acc0, false, false);
    acc1 = __builtin_amdgcn_wmma_f32_16x16x32_bf16(false, a, false, b1, (short)0, acc1, false, false);
  }
#pragma unroll
  for (int r = 0; r < 8; ++r) {
    int rr = m0 + r + 8 * hi;        // C/D layout: M = r + 8*hi
    int tt = rr / B_, bb = rr % B_;
    int c0i = n0 + lr, c1i = n0 + 16 + lr;
    xp[(((size_t)tt * G_ + g) * B_ + bb) * H_ + c0i] = acc0[r] + bx[g * H_ + c0i];
    xp[(((size_t)tt * G_ + g) * B_ + bb) * H_ + c1i] = acc1[r] + bx[g * H_ + c1i];
  }
}

// One timestep. Block owns (all 4 gates) x 16 batch rows x 64 hidden cols.
// WMMA loop over K=H, then fused gate nonlinearity + c/h update + next-step
// masked-bf16 h production (into the other hmb buffer: no cross-block races).
__global__ void lstm_step(int t,
                          const __bf16* __restrict__ hmb_in, __bf16* __restrict__ hmb_out,
                          const __bf16* __restrict__ Uhb,
                          const float* __restrict__ bh, const float* __restrict__ zh,
                          const float* __restrict__ xp, float* __restrict__ c,
                          float* __restrict__ hn) {
  __shared__ float gbuf[G_][16][64];
  constexpr int NB = H_ / 64;        // 16
  int bt   = blockIdx.x / NB;
  int nblk = blockIdx.x % NB;
  int b0 = bt * 16;
  int h0 = nblk * 64;
  int wave = threadIdx.x >> 5;
  int lane = threadIdx.x & 31;
  int hi = lane >> 4, lr = lane & 15;
  int g  = wave >> 1;                // 8 waves = 4 gates x 2 col-halves
  int n0 = h0 + (wave & 1) * 32;
  int brow = b0 + lr;
  const __bf16* ar = hmb_in + ((size_t)g * B_ + brow) * H_;
  v8f acc0 = {}, acc1 = {};
  for (int kb = 0; kb < H_; kb += 32) {
    v16bf a;
    const __bf16* ap = ar + kb + 16 * hi;
#pragma unroll
    for (int e = 0; e < 8; ++e) { a[e] = ap[e]; a[e + 8] = ap[8 + e]; }
    const v16bf bm0 = *(const v16bf*)(Uhb + ((size_t)g * H_ + (n0 + lr))      * H_ + kb + 16 * hi);
    const v16bf bm1 = *(const v16bf*)(Uhb + ((size_t)g * H_ + (n0 + 16 + lr)) * H_ + kb + 16 * hi);
    acc0 = __builtin_amdgcn_wmma_f32_16x16x32_bf16(false, a, false, bm0, (short)0, acc0, false, false);
    acc1 = __builtin_amdgcn_wmma_f32_16x16x32_bf16(false, a, false, bm1, (short)0, acc1, false, false);
  }
  const float* xpt = xp + (((size_t)t * G_ + g) * B_) * H_;
#pragma unroll
  for (int r = 0; r < 8; ++r) {
    int rb = r + 8 * hi;             // local batch row 0..15
    int bb = b0 + rb;
    int c0i = n0 + lr, c1i = n0 + 16 + lr;
    gbuf[g][rb][c0i - h0] = acc0[r] + bh[g * H_ + c0i] + xpt[(size_t)bb * H_ + c0i];
    gbuf[g][rb][c1i - h0] = acc1[r] + bh[g * H_ + c1i] + xpt[(size_t)bb * H_ + c1i];
  }
  __syncthreads();
  for (int idx = threadIdx.x; idx < 16 * 64; idx += blockDim.x) {
    int lrr = idx >> 6, lc = idx & 63;
    int bb = b0 + lrr, hh = h0 + lc;
    float gi = gbuf[0][lrr][lc], gf = gbuf[1][lrr][lc];
    float gc = gbuf[2][lrr][lc], go = gbuf[3][lrr][lc];
    float i_ = 1.f / (1.f + __expf(-gi));
    float f_ = 1.f / (1.f + __expf(-gf));
    float o_ = 1.f / (1.f + __expf(-go));
    float cn = f_ * c[(size_t)bb * H_ + hh] + i_ * tanhf(gc);
    float hw = o_ * tanhf(cn);
    c[(size_t)bb * H_ + hh] = cn;
    hn[((size_t)t * B_ + bb) * H_ + hh] = hw;
#pragma unroll
    for (int gg = 0; gg < G_; ++gg)
      hmb_out[((size_t)gg * B_ + bb) * H_ + hh] = (__bf16)(hw * zh[((size_t)gg * B_ + bb) * H_ + hh]);
  }
}

__global__ void finalize(const float* __restrict__ hn_last, const float* __restrict__ c,
                         float* __restrict__ hT, float* __restrict__ cT) {
  int i = blockIdx.x * blockDim.x + threadIdx.x;
  if (i < B_ * H_) { hT[i] = hn_last[i]; cT[i] = c[i]; }
}

extern "C" void kernel_launch(void* const* d_in, const int* in_sizes, int n_in,
                              void* d_out, int out_size, void* d_ws, size_t ws_size,
                              hipStream_t stream) {
  const float* x  = (const float*)d_in[0];
  const float* h0 = (const float*)d_in[1];
  const float* c0 = (const float*)d_in[2];
  const float* zx = (const float*)d_in[3];
  const float* zh = (const float*)d_in[4];
  const float* Wx = (const float*)d_in[5];
  const float* bx = (const float*)d_in[6];
  const float* Uh = (const float*)d_in[7];
  const float* bh = (const float*)d_in[8];

  char* ws = (char*)d_ws;
  __bf16* Wxb = (__bf16*)(ws + WXB_OFF);
  __bf16* Uhb = (__bf16*)(ws + UHB_OFF);
  __bf16* hmb = (__bf16*)(ws + HMB_OFF);   // [2][G][B][H]
  float*  cst = (float*)(ws + C_OFF);
  float*  xp  = (float*)(ws + XP_OFF);

  float* hn = (float*)d_out;
  float* hT = hn + (size_t)T_ * B_ * H_;
  float* cT = hT + (size_t)B_ * H_;

  { int n = G_ * H_ * I_; cvt_bf16<<<(n + 255) / 256, 256, 0, stream>>>(Wx, Wxb, n); }
  { int n = G_ * H_ * H_; cvt_bf16<<<(n + 255) / 256, 256, 0, stream>>>(Uh, Uhb, n); }
  { int n = G_ * B_ * H_; init_state<<<(n + 255) / 256, 256, 0, stream>>>(h0, c0, zh, hmb, cst); }

  xp_gemm<<<G_ * (T_ * B_ / 64) * (H_ / 64), 256, 0, stream>>>(x, zx, Wxb, bx, xp);

  size_t hsz = (size_t)G_ * B_ * H_;
  for (int t = 0; t < T_; ++t) {
    const __bf16* hin  = hmb + (size_t)(t & 1) * hsz;
    __bf16*       hout = hmb + (size_t)((t + 1) & 1) * hsz;
    lstm_step<<<(B_ / 16) * (H_ / 64), 256, 0, stream>>>(t, hin, hout, Uhb, bh, zh, xp, cst, hn);
  }

  { int n = B_ * H_; finalize<<<(n + 255) / 256, 256, 0, stream>>>(hn + (size_t)(T_ - 1) * B_ * H_, cst, hT, cT); }
}
